// Decoder_386547057101
// MI455X (gfx1250) — matrix operand order repaired and checked by reading, not compiled, not run on hardware
//
#include <hip/hip_runtime.h>

// Label-wise attention (CAML-style) on MI455X / gfx1250, wave32 + WMMA f16.
//   K1: z = tanh(x W^T + b)          -> f16 z [B][SP][DA] and zT [B][DA][S] in ws
//   K2: scores^T = z U^T             -> raw scores into alpha region of d_out
//       double-buffered pipeline: async global->LDS (ASYNCcnt) for z tiles,
//       register-staged fp32->f16 cvt for U tiles; s_wait_asynccnt 2 keeps
//       next tile's copies in flight while WMMAs consume the current tile.
//   K3: softmax over S, in place
//   K4: m^T = zT alpha^T (WMMA), fused y = V.m + Vb  (m never materialized)

#define B_   4
#define S_   4000
#define SP_  4096      // padded sequence rows for z (128-tile overrun safe)
#define DIN  1024
#define DA   512
#define L_   8921
#define KS   40        // LDS K-stride in f16 elems (80 B, 16B-aligned, bank-spread)

typedef __attribute__((ext_vector_type(16))) _Float16 v16h;
typedef __attribute__((ext_vector_type(8)))  _Float16 v8h;
typedef __attribute__((ext_vector_type(8)))  float    v8f;

static __device__ __forceinline__ v16h make_v16(v8h lo, v8h hi) {
  return __builtin_shufflevector(lo, hi, 0,1,2,3,4,5,6,7,8,9,10,11,12,13,14,15);
}

// A-fragment (16x32 f16, MxK) from LDS tile [rows][KS], rows m0..m0+15.
// Lanes 0-15 hold K=0..15, lanes 16-31 hold K=16..31 (element e -> K=h*16+e).
static __device__ __forceinline__ v16h frag_a(const _Float16* t, int m0, int lane) {
  const _Float16* p = t + (m0 + (lane & 15)) * KS;
  const int h = (lane >> 4) & 1;
  v8h lo = *(const v8h*)(p + h * 16);      // K = h*16 .. h*16+7
  v8h hi = *(const v8h*)(p + h * 16 + 8);  // K = h*16+8 ..
  return make_v16(lo, hi);
}

// B-fragment (32x16 f16, KxN) from LDS tile stored transposed [ncols][KS],
// cols n0..n0+15. Lanes 0-15: K=0..15, lanes 16-31: K=16..31 (element e -> K=h*16+e).
static __device__ __forceinline__ v16h frag_b(const _Float16* t, int n0, int lane) {
  const _Float16* p = t + (n0 + (lane & 15)) * KS + ((lane >> 4) & 1) * 16;
  v8h lo = *(const v8h*)(p);
  v8h hi = *(const v8h*)(p + 8);
  return make_v16(lo, hi);
}

#define WMMA(a, b, c) \
  __builtin_amdgcn_wmma_f32_16x16x32_f16(false, (a), false, (b), (short)0, (c), false, false)

// ---- gfx1250 async global->LDS copy (16 B per lane), ASYNCcnt tracked ------
// Generic LDS address: hardware uses addr[31:0] as the LDS offset (ISA 10.2).
static __device__ __forceinline__ void async_copy_b128(const void* gsrc, void* ldst) {
  unsigned lds_off = (unsigned)(unsigned long long)(uintptr_t)ldst;
  unsigned long long gaddr = (unsigned long long)(uintptr_t)gsrc;
  asm volatile("global_load_async_to_lds_b128 %0, %1, off"
               :: "v"(lds_off), "v"(gaddr) : "memory");
}
// Async loads complete in order: waiting ASYNCcnt<=2 after issuing 2 new copies
// guarantees the previous tile's 2 copies have landed in LDS.
static __device__ __forceinline__ void wait_async0() {
  asm volatile("s_wait_asynccnt 0x0" ::: "memory");
}
static __device__ __forceinline__ void wait_async2() {
  asm volatile("s_wait_asynccnt 0x2" ::: "memory");
}

// ---------------------------------------------------------------- K1: z, zT
__global__ __launch_bounds__(256) void k1_z(const float* __restrict__ x,
                                            const float* __restrict__ Ww,
                                            const float* __restrict__ Wb,
                                            _Float16* __restrict__ z,
                                            _Float16* __restrict__ zT) {
  __shared__ _Float16 As[128 * KS];   // x tile  [s][k]
  __shared__ _Float16 Bs[64 * KS];    // W rows  [a][k]  (== B^T layout)
  const int tid = threadIdx.x, lane = tid & 31, wave = tid >> 5;
  const int b = blockIdx.z;
  const int s0 = blockIdx.x * 128;
  const int a0 = blockIdx.y * 64;
  const int wm = (wave >> 1) * 32, wn = (wave & 1) * 32;
  const float* xb = x + (size_t)b * S_ * DIN;
  v8f acc[2][2] = {};
  for (int k0 = 0; k0 < DIN; k0 += 32) {
    for (int i = 0; i < 4; ++i) {                       // A: 128x32 fp32 -> f16
      int e = (i * 256 + tid) * 4;
      int r = e >> 5, c = e & 31;
      int s = s0 + r; if (s >= S_) s = S_ - 1;
      float4 v = *(const float4*)(xb + (size_t)s * DIN + k0 + c);
      _Float16* d = As + r * KS + c;
      d[0] = (_Float16)v.x; d[1] = (_Float16)v.y; d[2] = (_Float16)v.z; d[3] = (_Float16)v.w;
    }
    for (int i = 0; i < 2; ++i) {                       // B: 64x32 W rows
      int e = (i * 256 + tid) * 4;
      int r = e >> 5, c = e & 31;
      float4 v = *(const float4*)(Ww + (size_t)(a0 + r) * DIN + k0 + c);
      _Float16* d = Bs + r * KS + c;
      d[0] = (_Float16)v.x; d[1] = (_Float16)v.y; d[2] = (_Float16)v.z; d[3] = (_Float16)v.w;
    }
    __syncthreads();
    v16h af0 = frag_a(As, wm, lane),      af1 = frag_a(As, wm + 16, lane);
    v16h bf0 = frag_b(Bs, wn, lane),      bf1 = frag_b(Bs, wn + 16, lane);
    acc[0][0] = WMMA(af0, bf0, acc[0][0]);
    acc[0][1] = WMMA(af0, bf1, acc[0][1]);
    acc[1][0] = WMMA(af1, bf0, acc[1][0]);
    acc[1][1] = WMMA(af1, bf1, acc[1][1]);
    __syncthreads();
  }
  const int h = (lane >> 4) & 1, nl = lane & 15;
  _Float16* zb  = z  + (size_t)b * SP_ * DA;
  _Float16* ztb = zT + (size_t)b * DA * S_;
  for (int ti = 0; ti < 2; ++ti)
    for (int tj = 0; tj < 2; ++tj) {
      int sBase = s0 + wm + ti * 16 + h * 8;            // multiple of 8
      int a     = a0 + wn + tj * 16 + nl;
      if (sBase < S_) {                                  // whole 8-group in range
        float bias = Wb[a];
        v8h o;
        for (int r = 0; r < 8; ++r) {
          o[r] = (_Float16)tanhf(acc[ti][tj][r] + bias);
          zb[(size_t)(sBase + r) * DA + a] = o[r];
        }
        *(v8h*)(ztb + (size_t)a * S_ + sBase) = o;       // contiguous 16B
      }
    }
}

// ------------------------------------------------- K2: raw scores^T -> alpha
// Double-buffered: tile k+1 copies (async z -> LDS, U -> regs) overlap the
// WMMAs of tile k.  16 k-steps of 32 over DA=512.
__global__ __launch_bounds__(256) void k2_scores(const float* __restrict__ Uw,
                                                 const _Float16* __restrict__ z,
                                                 float* __restrict__ alpha) {
  __shared__ _Float16 As[2][128 * KS];  // z tile [s][k=a]    (2 x 10 KB)
  __shared__ _Float16 Bs[2][64 * KS];   // U rows [l][k=a]    (2 x 5 KB)
  const int tid = threadIdx.x, lane = tid & 31, wave = tid >> 5;
  const int b = blockIdx.z;
  const int s0 = blockIdx.x * 128;
  const int l0 = blockIdx.y * 64;
  const int wm = (wave >> 1) * 32, wn = (wave & 1) * 32;
  const _Float16* zb = z + (size_t)b * SP_ * DA;
  v8f acc[2][2] = {};

  // per-thread tile-copy coordinates (fixed across k-steps)
  const int ea0 = tid * 8,        ra0 = ea0 >> 5, ca0 = ea0 & 31;  // A chunk 0
  const int ea1 = (256 + tid) * 8, ra1 = ea1 >> 5, ca1 = ea1 & 31; // A chunk 1
  const int eb0 = tid * 4,        rb0 = eb0 >> 5, cb0 = eb0 & 31;  // B chunk 0
  const int eb1 = (256 + tid) * 4, rb1 = eb1 >> 5, cb1 = eb1 & 31; // B chunk 1
  int lb0 = l0 + rb0; if (lb0 >= L_) lb0 = L_ - 1;
  int lb1 = l0 + rb1; if (lb1 >= L_) lb1 = L_ - 1;

  // ---- prologue: stage k-step 0
  async_copy_b128(zb + (size_t)(s0 + ra0) * DA + ca0, &As[0][ra0 * KS + ca0]);
  async_copy_b128(zb + (size_t)(s0 + ra1) * DA + ca1, &As[0][ra1 * KS + ca1]);
  float4 bcur0 = *(const float4*)(Uw + (size_t)lb0 * DA + cb0);
  float4 bcur1 = *(const float4*)(Uw + (size_t)lb1 * DA + cb1);

  for (int kk = 0; kk < 16; ++kk) {
    const int cur = kk & 1, nxt = cur ^ 1;
    const int k1 = (kk + 1) * 32;
    // commit U tile k (cvt fp32 -> f16 into LDS)
    { _Float16* d = &Bs[cur][rb0 * KS + cb0];
      d[0]=(_Float16)bcur0.x; d[1]=(_Float16)bcur0.y; d[2]=(_Float16)bcur0.z; d[3]=(_Float16)bcur0.w; }
    { _Float16* d = &Bs[cur][rb1 * KS + cb1];
      d[0]=(_Float16)bcur1.x; d[1]=(_Float16)bcur1.y; d[2]=(_Float16)bcur1.z; d[3]=(_Float16)bcur1.w; }
    float4 bnxt0, bnxt1;
    if (kk < 15) {
      // stage k-step k+1: async z copies + U register loads
      async_copy_b128(zb + (size_t)(s0 + ra0) * DA + k1 + ca0, &As[nxt][ra0 * KS + ca0]);
      async_copy_b128(zb + (size_t)(s0 + ra1) * DA + k1 + ca1, &As[nxt][ra1 * KS + ca1]);
      bnxt0 = *(const float4*)(Uw + (size_t)lb0 * DA + k1 + cb0);
      bnxt1 = *(const float4*)(Uw + (size_t)lb1 * DA + k1 + cb1);
      wait_async2();                  // tile k landed; tile k+1 still in flight
    } else {
      wait_async0();
    }
    __syncthreads();
    v16h af0 = frag_a(As[cur], wm, lane),      af1 = frag_a(As[cur], wm + 16, lane);
    v16h bf0 = frag_b(Bs[cur], wn, lane),      bf1 = frag_b(Bs[cur], wn + 16, lane);
    acc[0][0] = WMMA(af0, bf0, acc[0][0]);
    acc[0][1] = WMMA(af0, bf1, acc[0][1]);
    acc[1][0] = WMMA(af1, bf0, acc[1][0]);
    acc[1][1] = WMMA(af1, bf1, acc[1][1]);
    __syncthreads();                  // buffers safe to overwrite next step
    bcur0 = bnxt0; bcur1 = bnxt1;
  }

  const int h = (lane >> 4) & 1, nl = lane & 15;
  float* ab = alpha + (size_t)b * L_ * S_;
  for (int ti = 0; ti < 2; ++ti)
    for (int tj = 0; tj < 2; ++tj) {
      int sBase = s0 + wm + ti * 16 + h * 8;
      int l     = l0 + wn + tj * 16 + nl;
      if (l < L_ && sBase < S_) {
        float* p = ab + (size_t)l * S_ + sBase;         // 32B aligned, contiguous s
        v8f a8 = acc[ti][tj];
        *(float4*)(p)     = make_float4(a8[0], a8[1], a8[2], a8[3]);
        *(float4*)(p + 4) = make_float4(a8[4], a8[5], a8[6], a8[7]);
      }
    }
}

// ------------------------------------------------------- K3: softmax over S
__global__ __launch_bounds__(256) void k3_softmax(float* __restrict__ alpha) {
  __shared__ float buf[S_];
  __shared__ float red[8];
  const int tid = threadIdx.x, lane = tid & 31, wave = tid >> 5;
  float* p = alpha + (size_t)blockIdx.x * S_;
  float mx = -3.4e38f;
  for (int i = tid; i < S_; i += 256) { float v = p[i]; buf[i] = v; mx = fmaxf(mx, v); }
  for (int o = 16; o > 0; o >>= 1) mx = fmaxf(mx, __shfl_xor(mx, o, 32));
  if (lane == 0) red[wave] = mx;
  __syncthreads();
  mx = red[0];
  for (int w = 1; w < 8; ++w) mx = fmaxf(mx, red[w]);
  __syncthreads();
  float sm = 0.f;
  for (int i = tid; i < S_; i += 256) { float e = __expf(buf[i] - mx); buf[i] = e; sm += e; }
  for (int o = 16; o > 0; o >>= 1) sm += __shfl_xor(sm, o, 32);
  if (lane == 0) red[wave] = sm;
  __syncthreads();
  float tot = 0.f;
  for (int w = 0; w < 8; ++w) tot += red[w];
  float inv = 1.0f / tot;
  for (int i = tid; i < S_; i += 256) p[i] = buf[i] * inv;
}

// ----------------------------------- K4: m^T = zT * alpha^T, fused y = V.m+b
__global__ __launch_bounds__(256) void k4_y(const float* __restrict__ Vw,
                                            const float* __restrict__ Vb,
                                            const _Float16* __restrict__ zT,
                                            const float* __restrict__ alpha,
                                            float* __restrict__ y) {
  __shared__ _Float16 As[DA * KS];    // zT tile [a][32 s]  (40 KB)
  __shared__ _Float16 Bs[16 * KS];    // alpha rows [l][32 s]
  __shared__ float ys[16];
  const int tid = threadIdx.x, lane = tid & 31, wave = tid >> 5;
  const int b = blockIdx.y, l0 = blockIdx.x * 16;
  const _Float16* ztb = zT + (size_t)b * DA * S_;
  const float* ab = alpha + (size_t)b * L_ * S_;
  v8f acc[4] = {};
  if (tid < 16) ys[tid] = 0.f;
  for (int s0 = 0; s0 < S_; s0 += 32) {                 // 125 exact iterations
    for (int i = 0; i < 8; ++i) {       // A: 512x32 f16 pure copy -> async to LDS
      int e = (i * 256 + tid) * 8;
      int r = e >> 5, c = e & 31;
      async_copy_b128(ztb + (size_t)r * S_ + s0 + c, As + r * KS + c);
    }
    for (int i = 0; i < 2; ++i) {       // B: 16x32 alpha fp32 -> f16 (VALU cvt)
      int e = i * 256 + tid;
      int r = e >> 5, c = e & 31;
      int l = l0 + r; if (l >= L_) l = L_ - 1;
      Bs[r * KS + c] = (_Float16)ab[(size_t)l * S_ + s0 + c];
    }
    wait_async0();
    __syncthreads();
    v16h bf = frag_b(Bs, 0, lane);
    for (int t = 0; t < 4; ++t) {
      v16h af = frag_a(As, wave * 64 + t * 16, lane);
      acc[t] = WMMA(af, bf, acc[t]);
    }
    __syncthreads();
  }
  const int h = (lane >> 4) & 1, nl = lane & 15;
  const int l = l0 + nl;
  float part = 0.f;
  if (l < L_) {
    for (int t = 0; t < 4; ++t) {
      const float* vrow = Vw + (size_t)l * DA + wave * 64 + t * 16 + h * 8;
      for (int r = 0; r < 8; ++r) part += acc[t][r] * vrow[r];  // contiguous 32B reads
    }
  }
  atomicAdd(&ys[nl], part);                              // ds_add_f32
  __syncthreads();
  if (tid < 16) {
    int ll = l0 + tid;
    if (ll < L_) y[(size_t)b * L_ + ll] = ys[tid] + Vb[ll];
  }
}

// --------------------------------------------------------------------------
extern "C" void kernel_launch(void* const* d_in, const int* in_sizes, int n_in,
                              void* d_out, int out_size, void* d_ws, size_t ws_size,
                              hipStream_t stream) {
  const float* x  = (const float*)d_in[0];
  const float* Ww = (const float*)d_in[1];
  const float* Wb = (const float*)d_in[2];
  const float* Uw = (const float*)d_in[3];
  const float* Vw = (const float*)d_in[4];
  const float* Vb = (const float*)d_in[5];
  float* y     = (float*)d_out;                 // [B, L]
  float* alpha = y + (size_t)B_ * L_;           // [B, L, S]
  _Float16* z  = (_Float16*)d_ws;               // [B][SP_][DA]  16.8 MB
  _Float16* zT = z + (size_t)B_ * SP_ * DA;     // [B][DA][S_]   16.4 MB

  k1_z     <<<dim3(32, DA / 64, B_),        256, 0, stream>>>(x, Ww, Wb, z, zT);
  k2_scores<<<dim3(32, (L_ + 63) / 64, B_), 256, 0, stream>>>(Uw, z, alpha);
  k3_softmax<<<dim3(B_ * L_),               256, 0, stream>>>(alpha);
  k4_y     <<<dim3((L_ + 15) / 16, B_),     256, 0, stream>>>(Vw, Vb, zT, alpha, y);
}
